// MultiHeadAttention_23235773071643
// MI455X (gfx1250) — matrix operand order repaired and checked by reading, not compiled, not run on hardware
//
#include <hip/hip_runtime.h>
#include <hip/hip_bf16.h>

// ---------------------------------------------------------------------------
// Fused MHA + residual + LayerNorm for MI455X (gfx1250, wave32, WMMA, TDM).
// Shapes: B=4, S=2048, D=512, H=8, dk=dv=64.
//
// Roofline: ~52 GFLOP total vs a mandatory >=0.6 GB of HBM traffic (537 MB of
// that is the attn tensor write) -> bandwidth-bound at 23.3 TB/s (~26us floor).
// Strategy: all GEMMs on the f16 WMMA path (f32 accum), attn tensor written
// exactly once from a fused mask+softmax+PV kernel whose score strip lives in
// LDS, K operand streamed into LDS with the Tensor Data Mover (double
// buffered, s_wait_tensorcnt), f16 intermediates to keep bytes small.
// ---------------------------------------------------------------------------

typedef __attribute__((ext_vector_type(16))) _Float16 v16h;
typedef __attribute__((ext_vector_type(8)))  _Float16 v8h;
typedef __attribute__((ext_vector_type(4)))  _Float16 v4h;
typedef __attribute__((ext_vector_type(8)))  float    v8f;
typedef __attribute__((ext_vector_type(4)))  float    v4f;
typedef __attribute__((ext_vector_type(4)))  unsigned int u32x4_t;
typedef __attribute__((ext_vector_type(4)))  int      i32x4_t;
typedef __attribute__((ext_vector_type(8)))  int      i32x8_t;

union F16x16 { v16h v; v8h h2[2]; _Float16 e[16]; };
union F32x8  { v8f v; float e[8]; };

static constexpr int kB  = 4;
static constexpr int kS  = 2048;
static constexpr int kD  = 512;
static constexpr int kH  = 8;
static constexpr int kBS = kB * kS;      // 8192 rows
static constexpr int kBH = kB * kH;      // 32 (b,h) pairs

// LDS layout of the attention kernel's dynamic shared blob (bytes):
static constexpr unsigned kScBytes    = 16 * kS * 4;            // 131072
static constexpr unsigned kRedOff     = kScBytes;               // +1024
static constexpr unsigned kOutOff     = kRedOff + 16 * 16 * 4;  // +4096
static constexpr unsigned kKStageOff  = kOutOff + 16 * 64 * 4;  // +2*16384
static constexpr unsigned kAttnLds    = kKStageOff + 2 * 128 * 64 * 2;

__device__ __forceinline__ v8f wmma_f32_f16(v16h a, v16h b, v8f c) {
  // 8 args: (neg_a, A, neg_b, B, c_mod, C, reuse_a, reuse_b)
  return __builtin_amdgcn_wmma_f32_16x16x32_f16(false, a, false, b, (short)0, c,
                                                false, false);
}

// A fragment (16x32 f16, [M,K] row-major). Lane l: row m0+(l&15);
// halves 0..7 = K k0+((l>>4)<<3)+0..7, halves 8..15 = +16 (two b128 loads).
__device__ __forceinline__ v16h load_frag_a_f16(const _Float16* base, int ld,
                                                int m0, int k0, int lane) {
  const int row = m0 + (lane & 15);
  const int ko  = (lane >> 4) << 3;
  F16x16 r;
  r.h2[0] = *(const v8h*)(base + (size_t)row * ld + k0 + ko);
  r.h2[1] = *(const v8h*)(base + (size_t)row * ld + k0 + 16 + ko);
  return r.v;
}

// B fragment (32x16 f16, source stored [N,K] row-major). Lane l: col n0+(l&15);
// halves 0..7 = K k0+((l>>4)<<3)+0..7, halves 8..15 = +16 (two b128 loads).
__device__ __forceinline__ v16h load_frag_b_f16(const _Float16* base, int ld,
                                                int n0, int k0, int lane) {
  const int row = n0 + (lane & 15);
  const int ko  = (lane >> 4) << 3;
  F16x16 r;
  r.h2[0] = *(const v8h*)(base + (size_t)row * ld + k0 + ko);
  r.h2[1] = *(const v8h*)(base + (size_t)row * ld + k0 + 16 + ko);
  return r.v;
}

// A fragment from an f32 source (LDS score strip), converting to f16.
__device__ __forceinline__ v16h load_frag_a_f32(const float* base, int ld,
                                                int m0, int k0, int lane) {
  const int row = m0 + (lane & 15);
  const int ko  = (lane >> 4) << 3;
  const float* p = base + (size_t)row * ld + k0 + ko;
  F16x16 r;
#pragma unroll
  for (int i = 0; i < 2; ++i) {
    v4f x0 = *(const v4f*)(p + 16 * i);
    v4f x1 = *(const v4f*)(p + 16 * i + 4);
    r.e[8 * i + 0] = (_Float16)x0[0];
    r.e[8 * i + 1] = (_Float16)x0[1];
    r.e[8 * i + 2] = (_Float16)x0[2];
    r.e[8 * i + 3] = (_Float16)x0[3];
    r.e[8 * i + 4] = (_Float16)x1[0];
    r.e[8 * i + 5] = (_Float16)x1[1];
    r.e[8 * i + 6] = (_Float16)x1[2];
    r.e[8 * i + 7] = (_Float16)x1[3];
  }
  return r.v;
}

// ---------------------------------------------------------------------------
// Tensor Data Mover: 1-D tile of f16 elements, global -> LDS.
// Descriptor packed per CDNA5 ISA:
//   group0: count=1 | lds_addr[63:32] | global_addr[120:64] | type=2[127:126]
//   group1: data_size=1(2B)[17:16], tensor_dim0[79:48], tensor_dim1[111:80]=1,
//           tile_dim0[127:112], tensor_dim0_stride[207:160]
// Arity differs by toolchain (clang-22: 5 args, clang-23: 6 args).
// ---------------------------------------------------------------------------
__device__ __forceinline__ void tdm_load_1d_f16(const _Float16* gsrc,
                                                unsigned ldsByteOff,
                                                unsigned tileElems,
                                                unsigned tensorElems) {
  const unsigned long long ga = (unsigned long long)(size_t)gsrc;
  u32x4_t g0;
  g0[0] = 1u;                                             // count=1, user mode
  g0[1] = ldsByteOff;                                     // lds_addr (bytes)
  g0[2] = (unsigned)(ga & 0xffffffffu);                   // global_addr[31:0]
  g0[3] = (unsigned)((ga >> 32) & 0x01ffffffu) | (2u << 30);  // addr[56:32]|type=2
  i32x8_t g1;
  g1[0] = (int)(1u << 16);                                // wg_mask=0, data_size=2B
  g1[1] = (int)((tensorElems & 0xffffu) << 16);           // tensor_dim0[15:0]
  g1[2] = (int)((tensorElems >> 16) & 0xffffu)            // tensor_dim0[31:16]
        | (int)(1u << 16);                                // tensor_dim1 = 1
  g1[3] = (int)((tileElems & 0xffffu) << 16);             // tile_dim0
  g1[4] = 0;                                              // tile_dim1/2 unused
  g1[5] = (int)tensorElems;                               // dim0_stride[31:0]
  g1[6] = 0;                                              // dim0_stride[47:32]
  g1[7] = 0;
  const i32x4_t z4 = {0, 0, 0, 0};
#if defined(__clang_major__) && (__clang_major__ >= 23)
  const i32x8_t z8 = {0, 0, 0, 0, 0, 0, 0, 0};
  __builtin_amdgcn_tensor_load_to_lds(g0, g1, z4, z4, z8, 0);
#else
  __builtin_amdgcn_tensor_load_to_lds(g0, g1, z4, z4, 0);
#endif
}

// ---------------------------------------------------------------------------
// Kernel 1: X[8192,512] @ W[512,512] -> f16, head-split layouts.
// mode 0: out[(b*H+h)*S + s][d]  (Q and K; [N,K]-friendly for QK^T)
// mode 1: out[(b*H+h)*64 + d][s] (V transposed; [N,K]-friendly for P@V)
// WG tile 128(M) x 64(N), 8 wave32s, each wave: 16-row strip x 4 n-tiles.
// ---------------------------------------------------------------------------
__global__ __launch_bounds__(256)
void gemm_qkv_kernel(const float* __restrict__ X, const float* __restrict__ W,
                     _Float16* __restrict__ out, int mode, float outScale) {
  __shared__ _Float16 Xs[128 * 32];   // [m][k] f16
  __shared__ _Float16 Ws[64 * 32];    // [n][k] f16 (transposed while staging)
  const int t = threadIdx.x, lane = t & 31, w = t >> 5;
  const int wgM = blockIdx.x * 128, wgN = blockIdx.y * 64;
  v8f acc[4] = {};

  for (int kb = 0; kb < 16; ++kb) {
    const int k0 = kb * 32;
    {  // stage X tile: thread t -> row t>>1, 16 cols at (t&1)*16
      const int row = t >> 1, cb = (t & 1) << 4;
      const float* src = X + (size_t)(wgM + row) * kD + k0 + cb;
      if (kb < 15) __builtin_prefetch(src + 32, 0, 1);  // next k-step
      _Float16* dst = &Xs[row * 32 + cb];
#pragma unroll
      for (int i = 0; i < 16; i += 4) {
        v4f xv = *(const v4f*)(src + i);
        v4h hv;
        hv[0] = (_Float16)xv[0]; hv[1] = (_Float16)xv[1];
        hv[2] = (_Float16)xv[2]; hv[3] = (_Float16)xv[3];
        *(v4h*)(dst + i) = hv;
      }
    }
    {  // stage W^T tile: thread t -> n = t>>2, 8 k's at (t&3)*8
      const int n = t >> 2, kc = (t & 3) << 3;
      v8h hv;
#pragma unroll
      for (int j = 0; j < 8; ++j)
        hv[j] = (_Float16)W[(size_t)(k0 + kc + j) * kD + wgN + n];
      *(v8h*)(&Ws[n * 32 + kc]) = hv;
    }
    __syncthreads();

    v16h a = load_frag_a_f16(Xs, 32, w * 16, 0, lane);
#pragma unroll
    for (int nt = 0; nt < 4; ++nt) {
      v16h bf = load_frag_b_f16(Ws, 32, nt * 16, 0, lane);
      acc[nt] = wmma_f32_f16(a, bf, acc[nt]);
    }
    __syncthreads();
  }

  // Epilogue: C 16x16 f32 layout -> lane l, reg r: m = r + ((l>>4)<<3), n = l&15
#pragma unroll
  for (int nt = 0; nt < 4; ++nt) {
    F32x8 c; c.v = acc[nt];
#pragma unroll
    for (int r = 0; r < 8; ++r) {
      const int m = wgM + w * 16 + r + ((lane >> 4) << 3);
      const int n = wgN + nt * 16 + (lane & 15);
      const int bb = m >> 11, s = m & (kS - 1);
      const int hh = n >> 6,  d = n & 63;
      const _Float16 val = (_Float16)(c.e[r] * outScale);
      if (mode == 0)
        out[(((size_t)(bb * kH + hh) * kS) + s) * 64 + d] = val;
      else
        out[(((size_t)(bb * kH + hh) * 64) + d) * kS + s] = val;
    }
  }
}

// ---------------------------------------------------------------------------
// Kernel 2: fused scores -> mask -> softmax -> attn write -> P@V.
// grid (S/16, B*H); 256 threads = 8 wave32s.
// Dynamic LDS: 16x2048 f32 scores + reduce + combine + 2x16KB K TDM buffers.
// K operand is streamed with tensor_load_to_lds: each jj step consumes a
// contiguous 128-row x 64-f16 strip of K (16KB), double buffered; wave 0
// issues the next descriptor while all waves run WMMA on the current one.
// ---------------------------------------------------------------------------
__global__ __launch_bounds__(256)
void attn_kernel(const _Float16* __restrict__ q16, const _Float16* __restrict__ k16,
                 const _Float16* __restrict__ vT16, const int* __restrict__ src_mask,
                 float* __restrict__ attn_out, float* __restrict__ attnV) {
  extern __shared__ char smem[];
  float* scTile  = (float*)smem;                        // [16][2048]
  float* red     = (float*)(smem + kRedOff);            // [16][16]
  float* outTile = (float*)(smem + kOutOff);            // [16][64]
  const _Float16* kStage = (const _Float16*)(smem + kKStageOff);  // 2x[128][64]

  const int t = threadIdx.x, lane = t & 31, w = t >> 5;
  const int qt = blockIdx.x, bh = blockIdx.y;
  const int b = bh >> 3, h = bh & 7;
  const int q0 = qt * 16;
  const _Float16* qBase = q16  + (size_t)bh * kS * 64;
  const _Float16* kBase = k16  + (size_t)bh * kS * 64;
  const _Float16* vBase = vT16 + (size_t)bh * 64 * kS;

  // Loop-invariant A fragments (Q rows q0..q0+15) and query-mask values.
  const v16h aq0 = load_frag_a_f16(qBase, 64, q0, 0, lane);
  const v16h aq1 = load_frag_a_f16(qBase, 64, q0, 32, lane);
  int mqv[8];
#pragma unroll
  for (int r = 0; r < 8; ++r)
    mqv[r] = src_mask[b * kS + q0 + r + ((lane >> 4) << 3)];

  // Prime the TDM pipeline with the first K strip.
  if (w == 0) tdm_load_1d_f16(kBase, kKStageOff, 128 * 64, kS * 64);

  // ---- Phase 1: scores S = (Q/sqrt(dk)) K^T ----
  for (int jj = 0; jj < 16; ++jj) {
    if (w == 0) {
      if (jj + 1 < 16) {
        tdm_load_1d_f16(kBase + (size_t)(jj + 1) * 128 * 64,
                        kKStageOff + (unsigned)((jj + 1) & 1) * 16384u,
                        128 * 64, kS * 64);
        __builtin_amdgcn_s_wait_tensorcnt(1);  // strip jj landed
      } else {
        __builtin_amdgcn_s_wait_tensorcnt(0);
      }
    }
    __syncthreads();  // strip jj visible to all waves

    const _Float16* kt = kStage + (size_t)(jj & 1) * (128 * 64);
    v8f acc = {};
    v16h b0 = load_frag_b_f16(kt, 64, w * 16, 0, lane);
    acc = wmma_f32_f16(aq0, b0, acc);
    v16h b1 = load_frag_b_f16(kt, 64, w * 16, 32, lane);
    acc = wmma_f32_f16(aq1, b1, acc);

    const int j  = jj * 8 + w;            // key tile 0..127
    const int n  = j * 16 + (lane & 15);
    const int mk = src_mask[b * kS + n];
    F32x8 c; c.v = acc;
#pragma unroll
    for (int r = 0; r < 8; ++r) {
      const int m  = r + ((lane >> 4) << 3);
      const int qg = q0 + m;
      float val = c.e[r];
      if (((mqv[r] & mk) == 0) && (qg != n)) val = -1e-9f;  // faithful MASK_FILL
      scTile[m * kS + n] = val;
    }
    __syncthreads();  // reads of buffer jj&1 done before it is refilled
  }

  // ---- Phase 2: softmax over rows of 2048 (16 threads/row, v4f interleave
  //      so lanes store consecutive 16B chunks -> coalesced b128 stores) ----
  {
    const int row = t >> 4, seg = t & 15;
    float* rp = scTile + row * kS;
    float mx = -3.0e38f;
    for (int i = 0; i < 32; ++i) {
      v4f x = *(const v4f*)(rp + (seg + i * 16) * 4);
      mx = fmaxf(mx, fmaxf(fmaxf(x[0], x[1]), fmaxf(x[2], x[3])));
    }
    red[row * 16 + seg] = mx;
    __syncthreads();
    float m2 = -3.0e38f;
    for (int j = 0; j < 16; ++j) m2 = fmaxf(m2, red[row * 16 + j]);
    __syncthreads();
    float ssum = 0.f;
    for (int i = 0; i < 32; ++i) {
      float* p = rp + (seg + i * 16) * 4;
      v4f x = *(const v4f*)p;
      x[0] = __expf(x[0] - m2); x[1] = __expf(x[1] - m2);
      x[2] = __expf(x[2] - m2); x[3] = __expf(x[3] - m2);
      *(v4f*)p = x;
      ssum += (x[0] + x[1]) + (x[2] + x[3]);
    }
    red[row * 16 + seg] = ssum;
    __syncthreads();
    float tot = 0.f;
    for (int j = 0; j < 16; ++j) tot += red[row * 16 + j];
    const float inv = 1.0f / tot;
    float* aout = attn_out + ((size_t)bh * kS + (q0 + row)) * kS;
    for (int i = 0; i < 32; ++i) {
      const int c0 = (seg + i * 16) * 4;
      v4f x = *(const v4f*)(rp + c0);
      x[0] *= inv; x[1] *= inv; x[2] *= inv; x[3] *= inv;
      *(v4f*)(rp + c0) = x;              // keep normalized P for P@V
      *(v4f*)(aout + c0) = x;            // single write of the 537MB attn tensor
    }
  }
  for (int i = t; i < 16 * 64; i += 256) outTile[i] = 0.f;
  __syncthreads();

  // ---- Phase 3: O = P @ V. Wave w: n-tile w&3, K-half w>>2; ds_add combine ----
  {
    const int nt = w & 3, khalf = w >> 2;
    v8f acc = {};
    for (int ks = khalf * 32; ks < khalf * 32 + 32; ++ks) {
      v16h a  = load_frag_a_f32(scTile, kS, 0, ks * 32, lane);
      v16h bf = load_frag_b_f16(vBase, kS, nt * 16, ks * 32, lane);
      acc = wmma_f32_f16(a, bf, acc);
    }
    F32x8 c; c.v = acc;
#pragma unroll
    for (int r = 0; r < 8; ++r) {
      const int m = r + ((lane >> 4) << 3);
      const int n = nt * 16 + (lane & 15);
      atomicAdd(&outTile[m * 64 + n], c.e[r]);  // ds_add_f32
    }
  }
  __syncthreads();

  for (int i = t; i < 16 * 64; i += 256) {
    const int m = i >> 6, d = i & 63;
    attnV[((size_t)(b * kS) + q0 + m) * kD + h * 64 + d] = outTile[i];
  }
}

// ---------------------------------------------------------------------------
// Kernel 3: proj = attnV[8192,512] @ Wo[512,512] + residual (f16 WMMA path).
// ---------------------------------------------------------------------------
__global__ __launch_bounds__(256)
void gemm_out_kernel(const float* __restrict__ X, const float* __restrict__ W,
                     const float* __restrict__ resid, float* __restrict__ proj) {
  __shared__ _Float16 Xs[128 * 32];
  __shared__ _Float16 Ws[64 * 32];
  const int t = threadIdx.x, lane = t & 31, w = t >> 5;
  const int wgM = blockIdx.x * 128, wgN = blockIdx.y * 64;
  v8f acc[4] = {};

  for (int kb = 0; kb < 16; ++kb) {
    const int k0 = kb * 32;
    {
      const int row = t >> 1, cb = (t & 1) << 4;
      const float* src = X + (size_t)(wgM + row) * kD + k0 + cb;
      if (kb < 15) __builtin_prefetch(src + 32, 0, 1);
      _Float16* dst = &Xs[row * 32 + cb];
#pragma unroll
      for (int i = 0; i < 16; i += 4) {
        v4f xv = *(const v4f*)(src + i);
        v4h hv;
        hv[0] = (_Float16)xv[0]; hv[1] = (_Float16)xv[1];
        hv[2] = (_Float16)xv[2]; hv[3] = (_Float16)xv[3];
        *(v4h*)(dst + i) = hv;
      }
    }
    {
      const int n = t >> 2, kc = (t & 3) << 3;
      v8h hv;
#pragma unroll
      for (int j = 0; j < 8; ++j)
        hv[j] = (_Float16)W[(size_t)(k0 + kc + j) * kD + wgN + n];
      *(v8h*)(&Ws[n * 32 + kc]) = hv;
    }
    __syncthreads();
    v16h a = load_frag_a_f16(Xs, 32, w * 16, 0, lane);
#pragma unroll
    for (int nt = 0; nt < 4; ++nt) {
      v16h bf = load_frag_b_f16(Ws, 32, nt * 16, 0, lane);
      acc[nt] = wmma_f32_f16(a, bf, acc[nt]);
    }
    __syncthreads();
  }

#pragma unroll
  for (int nt = 0; nt < 4; ++nt) {
    F32x8 c; c.v = acc[nt];
#pragma unroll
    for (int r = 0; r < 8; ++r) {
      const int m = wgM + w * 16 + r + ((lane >> 4) << 3);
      const int n = wgN + nt * 16 + (lane & 15);
      proj[(size_t)m * kD + n] = c.e[r] + resid[(size_t)m * kD + n];
    }
  }
}

// ---------------------------------------------------------------------------
// Kernel 4: LayerNorm over D=512, one block per row, 2 elems/thread.
// ---------------------------------------------------------------------------
__global__ __launch_bounds__(256)
void ln_kernel(const float* __restrict__ proj, const float* __restrict__ gamma,
               const float* __restrict__ beta, float* __restrict__ out) {
  __shared__ float rbuf[256];
  const int row = blockIdx.x, t = threadIdx.x;
  const float* rp = proj + (size_t)row * kD;
  const float x0 = rp[t], x1 = rp[t + 256];

  rbuf[t] = x0 + x1;
  __syncthreads();
  for (int off = 128; off > 0; off >>= 1) {
    if (t < off) rbuf[t] += rbuf[t + off];
    __syncthreads();
  }
  const float mu = rbuf[0] * (1.0f / kD);
  __syncthreads();

  const float d0 = x0 - mu, d1 = x1 - mu;
  rbuf[t] = d0 * d0 + d1 * d1;
  __syncthreads();
  for (int off = 128; off > 0; off >>= 1) {
    if (t < off) rbuf[t] += rbuf[t + off];
    __syncthreads();
  }
  const float rs = rsqrtf(rbuf[0] * (1.0f / kD) + 1e-6f);

  out[(size_t)row * kD + t]       = d0 * rs * gamma[t] + beta[t];
  out[(size_t)row * kD + t + 256] = d1 * rs * gamma[t + 256] + beta[t + 256];
}

// ---------------------------------------------------------------------------
extern "C" void kernel_launch(void* const* d_in, const int* in_sizes, int n_in,
                              void* d_out, int out_size, void* d_ws, size_t ws_size,
                              hipStream_t stream) {
  (void)in_sizes; (void)n_in; (void)out_size; (void)ws_size;
  const float* q        = (const float*)d_in[0];
  const float* k        = (const float*)d_in[1];
  const float* v        = (const float*)d_in[2];
  const int*   src_mask = (const int*)d_in[3];
  const float* Wq       = (const float*)d_in[4];
  const float* Wk       = (const float*)d_in[5];
  const float* Wv       = (const float*)d_in[6];
  const float* Wo       = (const float*)d_in[7];
  const float* gamma    = (const float*)d_in[8];
  const float* beta     = (const float*)d_in[9];

  float* outMain = (float*)d_out;                              // [B,S,D]
  float* outAttn = outMain + (size_t)kBS * kD;                 // [B,H,S,S]

  char* ws = (char*)d_ws;                                      // 256B-aligned regions
  _Float16* q16   = (_Float16*)(ws);                           //  8.4 MB
  _Float16* k16   = (_Float16*)(ws + (size_t)8388608);         //  8.4 MB
  _Float16* vT16  = (_Float16*)(ws + (size_t)16777216);        //  8.4 MB
  float*    attnV = (float*)(ws + (size_t)25165824);           // 16.8 MB
  float*    proj  = (float*)(ws + (size_t)41943040);           // 16.8 MB

  const dim3 blk(256);
  const dim3 gProj(kBS / 128, kD / 64);                        // 64 x 8

  // 1/sqrt(dk) folded into the Q projection output.
  gemm_qkv_kernel<<<gProj, blk, 0, stream>>>(q, Wq, q16, 0, 0.125f);
  gemm_qkv_kernel<<<gProj, blk, 0, stream>>>(k, Wk, k16, 0, 1.0f);
  gemm_qkv_kernel<<<gProj, blk, 0, stream>>>(v, Wv, vT16, 1, 1.0f);

  attn_kernel<<<dim3(kS / 16, kBH), blk, kAttnLds, stream>>>(
      q16, k16, vT16, src_mask, outAttn, attnV);

  gemm_out_kernel<<<gProj, blk, 0, stream>>>(attnV, Wo, q, proj);
  ln_kernel<<<dim3(kBS), blk, 0, stream>>>(proj, gamma, beta, outMain);
}
